// Attention_10385230922032
// MI455X (gfx1250) — matrix operand order repaired and checked by reading, not compiled, not run on hardware
//
#include <hip/hip_runtime.h>
#include <hip/hip_bf16.h>

// ---- CDNA5 (gfx1250) flash attention: causal, B=8, S=2048, D=128, fp32 io ----
typedef __attribute__((ext_vector_type(16))) __bf16        v16bf;
typedef __attribute__((ext_vector_type(8)))  float         v8f;
typedef __attribute__((ext_vector_type(4))) unsigned int   u32x4;
typedef __attribute__((ext_vector_type(2))) unsigned int   u32x2;
typedef __attribute__((ext_vector_type(4)))  float         f32x4;

#define BATCH 8
#define SEQ   2048
#define DM    128
#define QBLK  64
#define KVBLK 32
// 1/sqrt(128) * log2(e), folded into the Q tile at load time so scores land
// directly in the exp2 domain (no per-block v_mul on the score tiles).
#define SCALE_LOG2E (0.08838834764831845f * 1.4426950408889634f)

// LDS strides (halfwords): row*stride*2 bytes is a multiple of 16 (legal b128
// ds loads) and 16 consecutive lanes land in 16 distinct banks.
#define KSTRIDE 136   // 32 keys x 128 dk bf16, padded
#define VSTRIDE 40    // 128 dv rows x 32 kv bf16 (transposed, K-permuted), padded
#define PSTRIDE 40    // per-wave 16 q x 32 kv bf16 (K-permuted), padded

// ds_swizzle_b32 immediate XOR patterns (and_mask=0x1f, or=0, xor in [14:10]);
// XOR of 1/2/4/8 stays inside each 16-lane half -> row-wise max reduction with
// no index-setup VALU.
#define SWZ_F(x, pat) __builtin_bit_cast(float,                                 \
    __builtin_amdgcn_ds_swizzle(__builtin_bit_cast(int, (x)), (pat)))
#define RED16_MAX(v)                                                            \
  do {                                                                          \
    v = fmaxf(v, SWZ_F(v, 0x041F));                                             \
    v = fmaxf(v, SWZ_F(v, 0x081F));                                             \
    v = fmaxf(v, SWZ_F(v, 0x101F));                                             \
    v = fmaxf(v, SWZ_F(v, 0x201F));                                             \
  } while (0)

__device__ __forceinline__ unsigned short f2bf(float x) {
  __bf16 h = (__bf16)x;
  return __builtin_bit_cast(unsigned short, h);
}
__device__ __forceinline__ unsigned int pk2(float a, float b) {
  return (unsigned int)f2bf(a) | ((unsigned int)f2bf(b) << 16);
}

union V16 { v16bf bf; u32x4 q[2]; };

__global__ __launch_bounds__(128)
void fa_causal_kernel(const float* __restrict__ Qp,
                      const float* __restrict__ Kp,
                      const float* __restrict__ Vp,
                      float* __restrict__ Op) {
  __shared__ __align__(16) unsigned short Ksh[KVBLK * KSTRIDE];   //  8.7 KB
  __shared__ __align__(16) unsigned short Vsh[DM * VSTRIDE];      // 10.2 KB (V^T)
  __shared__ __align__(16) unsigned short Psh[4 * 16 * PSTRIDE];  //  5.1 KB

  const int tid   = threadIdx.x;
  const int wave  = tid >> 5;
  const int lane  = tid & 31;
  const int half  = lane >> 4;    // 16-lane half of the wave
  const int m     = lane & 15;    // column / row index within a 16-tile
  const int bidx  = blockIdx.x / (SEQ / QBLK);
  const int q0    = (blockIdx.x % (SEQ / QBLK)) * QBLK;
  const int qbase = q0 + wave * 16;
  const int pbase = wave * 16 * PSTRIDE;

  // ---- Q tile (16x128) -> registers, A-operand layout, pre-scaled ----
  // A layout (16-bit, 16x32): per 32-k step this lane holds
  // dk = s*32 + half*8 + [0..7] and dk = s*32 + 16 + half*8 + [0..7].
  V16 aQ[4];
  {
    const float* Qrow = Qp + ((size_t)bidx * SEQ + qbase + m) * DM;
#pragma unroll
    for (int s4 = 0; s4 < 4; ++s4) {
      const int d0 = s4 * 32 + half * 8;
      f32x4 f0 = *(const f32x4*)(Qrow + d0);
      f32x4 f1 = *(const f32x4*)(Qrow + d0 + 4);
      f32x4 f2 = *(const f32x4*)(Qrow + d0 + 16);
      f32x4 f3 = *(const f32x4*)(Qrow + d0 + 20);
      const float sc = SCALE_LOG2E;
      u32x4 a;
      a.x = pk2(f0.x * sc, f0.y * sc); a.y = pk2(f0.z * sc, f0.w * sc);
      a.z = pk2(f1.x * sc, f1.y * sc); a.w = pk2(f1.z * sc, f1.w * sc);
      aQ[s4].q[0] = a;
      a.x = pk2(f2.x * sc, f2.y * sc); a.y = pk2(f2.z * sc, f2.w * sc);
      a.z = pk2(f3.x * sc, f3.y * sc); a.w = pk2(f3.z * sc, f3.w * sc);
      aQ[s4].q[1] = a;
    }
  }

  // bf16 all-ones B operand: Lacc = wmma(P, ones, Lacc) gives the row sums of
  // P in C layout for free (every output column equals the row sum).
  V16 onesB;
  {
    u32x4 o; o.x = o.y = o.z = o.w = 0x3F803F80u;   // bf16 1.0 pair
    onesB.q[0] = o; onesB.q[1] = o;
  }

  // ---- Online-softmax state.  C layout: VGPR r holds row (r + 8*half). ----
  v8f Oacc[8];
#pragma unroll
  for (int d = 0; d < 8; ++d) Oacc[d] = (v8f){0.f,0.f,0.f,0.f,0.f,0.f,0.f,0.f};
  v8f Lacc = (v8f){0.f,0.f,0.f,0.f,0.f,0.f,0.f,0.f};  // softmax denominators
  float m_i[8];
#pragma unroll
  for (int r = 0; r < 8; ++r) m_i[r] = -3.0e38f;

  const int kend = q0 + QBLK;  // causal bound for this 64-row q block

  for (int kb = 0; kb < kend; kb += KVBLK) {
    __syncthreads();  // previous iteration's LDS reads complete

    const float* Kg = Kp + ((size_t)bidx * SEQ + kb) * DM;
    const float* Vg = Vp + ((size_t)bidx * SEQ + kb) * DM;
    if (kb + KVBLK < kend) {  // gfx1250 global_prefetch_b8 (warm GL2)
      __builtin_prefetch(Kg + KVBLK * DM, 0, 0);
      __builtin_prefetch(Vg + KVBLK * DM, 0, 0);
    }

    // ---- Cooperative stage: K as [key][dk] bf16; V^T as [dv][kappa] bf16
    // where the PV GEMM's K-dimension is permuted: kappa(kv) = 2*(kv&15)|(kv>>4)
    // (same permutation applied to P below, so the product is unchanged).
#pragma unroll
    for (int i = 0; i < 8; ++i) {
      const int f  = tid + i * 128;     // 1024 float4s over 128 threads
      const int kv = f >> 5;
      const int c4 = f & 31;
      f32x4 kq = *(const f32x4*)(Kg + kv * DM + c4 * 4);
      u32x2 kp; kp.x = pk2(kq.x, kq.y); kp.y = pk2(kq.z, kq.w);
      *(u32x2*)(&Ksh[kv * KSTRIDE + c4 * 4]) = kp;
      f32x4 vq = *(const f32x4*)(Vg + kv * DM + c4 * 4);
      const int dv = c4 * 4;
      const int kap = ((kv & 15) << 1) | (kv >> 4);
      Vsh[(dv + 0) * VSTRIDE + kap] = f2bf(vq.x);
      Vsh[(dv + 1) * VSTRIDE + kap] = f2bf(vq.y);
      Vsh[(dv + 2) * VSTRIDE + kap] = f2bf(vq.z);
      Vsh[(dv + 3) * VSTRIDE + kap] = f2bf(vq.w);
    }
    __syncthreads();

    // Causal: this wave needs this KV block only if kb <= qbase+15.
    if (kb >= qbase + 16) continue;   // still meets the top-of-loop barrier

    // ---- S = Q.K^T.  Pre-load ALL B operands so the ds_loads pipeline,
    // then run two independent accumulation chains (t=0 / t=1) interleaved.
    v8f sT[2];
    {
      V16 kB[2][4];
#pragma unroll
      for (int t = 0; t < 2; ++t)
#pragma unroll
        for (int s4 = 0; s4 < 4; ++s4) {
          const unsigned short* kp =
              &Ksh[(t * 16 + m) * KSTRIDE + s4 * 32 + half * 8];
          kB[t][s4].q[0] = *(const u32x4*)(kp);
          kB[t][s4].q[1] = *(const u32x4*)(kp + 16);
        }
      sT[0] = (v8f){0.f,0.f,0.f,0.f,0.f,0.f,0.f,0.f};
      sT[1] = (v8f){0.f,0.f,0.f,0.f,0.f,0.f,0.f,0.f};
#pragma unroll
      for (int s4 = 0; s4 < 4; ++s4)
#pragma unroll
        for (int t = 0; t < 2; ++t)
          sT[t] = __builtin_amdgcn_wmma_f32_16x16x32_bf16(
              false, aQ[s4].bf, false, kB[t][s4].bf, (short)0, sT[t],
              false, false);
    }

    const bool needmask = (kb + KVBLK - 1) > qbase;

    // ---- Online softmax in C layout; ds_swizzle row-max reductions ----
    float alpha[8];
    float mind = 0.0f;   // min_r (m_old - m_new); < 0 iff any row max moved
#pragma unroll
    for (int r = 0; r < 8; ++r) {
      const int qrow = qbase + r + 8 * half;
      float a0 = sT[0][r];                   // already in exp2 domain
      float a1 = sT[1][r];
      if (needmask) {
        a0 = (kb + m      > qrow) ? -3.0e38f : a0;
        a1 = (kb + 16 + m > qrow) ? -3.0e38f : a1;
      }
      float rm = fmaxf(a0, a1);
      RED16_MAX(rm);
      const float mnew = fmaxf(m_i[r], rm);
      const float dm   = m_i[r] - mnew;      // 0 if unchanged, < 0 otherwise
      mind     = fminf(mind, dm);
      alpha[r] = __builtin_amdgcn_exp2f(dm);
      const float p0 = __builtin_amdgcn_exp2f(a0 - mnew);
      const float p1 = __builtin_amdgcn_exp2f(a1 - mnew);
      m_i[r] = mnew;

      // Spill P to per-wave LDS in K-permuted order: p0 (kv=m) and p1
      // (kv=m+16) are adjacent halfwords -> single b32 store, 16 consecutive
      // banks across the half-wave.
      const int prow = r + 8 * half;
      *(unsigned int*)(&Psh[pbase + prow * PSTRIDE + 2 * m]) = pk2(p0, p1);
    }

    // Rescale O/L only when some row's max actually moved (rare in steady
    // state: alpha == 1.0 exactly when the max is unchanged).  __any is a
    // ballot -> scalar branch: wave-uniform, EXEC stays all-ones (WMMA-safe).
    if (__any(mind < 0.0f)) {
#pragma unroll
      for (int r = 0; r < 8; ++r) {
        Lacc[r] *= alpha[r];
#pragma unroll
        for (int d = 0; d < 8; ++d) Oacc[d][r] *= alpha[r];
      }
    }

    // wave-private LDS RAW fence (LDS in-order within a wave; belt+braces)
    asm volatile("s_wait_dscnt 0" ::: "memory");

    // ---- O += P.V and L += P.1 : 2-deep software pipeline on the vB
    // operands.  sched_barrier(0) pins the issue order load(d+1) -> wmma(d)
    // so the pre-RA scheduler cannot sink the loads below the WMMA (which
    // previously let RA coalesce both buffers into one and re-serialized the
    // chain).  With the order pinned, the wait before wmma(d) leaves tile
    // d+1's pair in flight (s_wait_dscnt 0x2): one WMMA of latency hiding.
    V16 pA;
    {
      const unsigned short* pp = &Psh[pbase + m * PSTRIDE + half * 16];
      pA.q[0] = *(const u32x4*)(pp);
      pA.q[1] = *(const u32x4*)(pp + 8);    // kappa +8, contiguous (matches V^T B)
    }
    V16 vb[2];
    {
      const unsigned short* vp = &Vsh[(0 * 16 + m) * VSTRIDE + half * 16];
      vb[0].q[0] = *(const u32x4*)(vp);
      vb[0].q[1] = *(const u32x4*)(vp + 8);
    }
    Lacc = __builtin_amdgcn_wmma_f32_16x16x32_bf16(
        false, pA.bf, false, onesB.bf, (short)0, Lacc, false, false);
#pragma unroll
    for (int d = 0; d < 8; ++d) {
      if (d < 7) {
        const unsigned short* vp =
            &Vsh[((d + 1) * 16 + m) * VSTRIDE + half * 16];
        vb[(d + 1) & 1].q[0] = *(const u32x4*)(vp);
        vb[(d + 1) & 1].q[1] = *(const u32x4*)(vp + 8);
      }
      __builtin_amdgcn_sched_barrier(0);   // keep load(d+1) above wmma(d)
      Oacc[d] = __builtin_amdgcn_wmma_f32_16x16x32_bf16(
          false, pA.bf, false, vb[d & 1].bf, (short)0, Oacc[d], false, false);
    }
  }

  // ---- Epilogue: O /= l, store fp32 ----
  float* Ob = Op + (size_t)bidx * SEQ * DM;
#pragma unroll
  for (int r = 0; r < 8; ++r) {
    const float inv  = 1.0f / Lacc[r];
    const int   qrow = qbase + r + 8 * half;
#pragma unroll
    for (int d = 0; d < 8; ++d)
      Ob[(size_t)qrow * DM + d * 16 + m] = Oacc[d][r] * inv;
  }
}

extern "C" void kernel_launch(void* const* d_in, const int* in_sizes, int n_in,
                              void* d_out, int out_size, void* d_ws, size_t ws_size,
                              hipStream_t stream) {
  (void)in_sizes; (void)n_in; (void)out_size; (void)d_ws; (void)ws_size;
  const float* Q = (const float*)d_in[0];
  const float* K = (const float*)d_in[1];
  const float* V = (const float*)d_in[2];
  // d_in[3] (additive causal mask) is applied analytically in-kernel.
  float* O = (float*)d_out;
  dim3 grid(BATCH * (SEQ / QBLK));
  dim3 block(128);
  fa_causal_kernel<<<grid, block, 0, stream>>>(Q, K, V, O);
}
